// MaritimeGNNTracker_52381421142047
// MI455X (gfx1250) — matrix operand order repaired and checked by reading, not compiled, not run on hardware
//
#include <hip/hip_runtime.h>
#include <math.h>

#define N_NODES 50000
#define N_EDGES 800000
#define SEQ     10
#define IN_DIM  8
#define HID     64
#define SP      32
#define BN_EPS  1e-5f

typedef __attribute__((ext_vector_type(16))) _Float16 v16h;
typedef __attribute__((ext_vector_type(8)))  float    v8f;

__device__ inline float sigf(float x) { return 1.0f / (1.0f + expf(-x)); }

__device__ inline void atomAddF(float* p, float v) {
    __hip_atomic_fetch_add(p, v, __ATOMIC_RELAXED, __HIP_MEMORY_SCOPE_AGENT);
}

// Pack 8 consecutive f32 (two float4 b128 loads) into fragment slots [base..base+7]
__device__ inline void cvt8(v16h& f, int base, const float4 a, const float4 b) {
    f[base + 0] = (_Float16)a.x; f[base + 1] = (_Float16)a.y;
    f[base + 2] = (_Float16)a.z; f[base + 3] = (_Float16)a.w;
    f[base + 4] = (_Float16)b.x; f[base + 5] = (_Float16)b.y;
    f[base + 6] = (_Float16)b.z; f[base + 7] = (_Float16)b.w;
}

// ---------------------------------------------------------------------------
// Generic WMMA GEMM: out[M,N] = act( A[M,K](lda,aoff) * W[N,K]^T + bias + resid )
// One wave per 16x16 output tile; f16 operands, f32 accumulate via
// v_wmma_f32_16x16x32_f16. M is a multiple of 16 at every call site.
// Fast path: full 32-K chunk + 16B-aligned rows -> b128 (float4) operand loads.
// N/K tails use guarded scalar loads; control flow re-converges before WMMA.
// ---------------------------------------------------------------------------
template <int ACT>  // 0 = none, 1 = relu, 2 = sigmoid
__global__ void __launch_bounds__(128) gemm_wmma_kernel(
    const float* __restrict__ A, long lda, long aoff,
    const float* __restrict__ W,      // [N,K] row-major
    const float* __restrict__ bias,   // [N] or nullptr
    const float* __restrict__ resid,  // [M,N] or nullptr
    float* __restrict__ out,          // [M,N]
    int M, int N, int K)
{
    const int lane = threadIdx.x & 31;
    const int wave = blockIdx.x * (blockDim.x >> 5) + (threadIdx.x >> 5);
    const int nT16 = (N + 15) >> 4;
    const int mT = wave / nT16;
    const int nT = wave - mT * nT16;
    if (mT * 16 >= M) return;  // wave-uniform

    const int mrow = mT * 16 + (lane & 15);  // A row owned by this lane
    const int ncol = nT * 16 + (lane & 15);  // B column / D column for this lane
    const int hi   = lane >> 4;              // lane group 0/1

    v8f acc;
    float binit = (bias != nullptr && ncol < N) ? bias[ncol] : 0.0f;
#pragma unroll
    for (int v = 0; v < 8; ++v) acc[v] = binit;
    if (resid != nullptr) {
#pragma unroll
        for (int v = 0; v < 8; ++v) {
            int m = mT * 16 + v + hi * 8;
            if (ncol < N) acc[v] += resid[(long)m * N + ncol];
        }
    }

    const long abase = (long)mrow * lda + aoff;
    const bool alignA = ((lda & 3) == 0) && ((aoff & 3) == 0);
    const bool alignW = ((K & 3) == 0);

    for (int k0 = 0; k0 < K; k0 += 32) {
        if (k0 + 32 < K) __builtin_prefetch(&A[abase + k0 + 32], 0, 0);
        const bool full = (k0 + 32 <= K);
        v16h af, bf;

        // ---- A fragment: lanes 0-15 hold K = k0+[0..7, 16..23]; lanes 16-31 +8
        if (full && alignA) {
            const float* pa = A + abase + k0 + hi * 8;
            float4 a0 = *(const float4*)(pa);
            float4 a1 = *(const float4*)(pa + 4);
            float4 a2 = *(const float4*)(pa + 16);
            float4 a3 = *(const float4*)(pa + 20);
            cvt8(af, 0, a0, a1);
            cvt8(af, 8, a2, a3);
        } else {
#pragma unroll
            for (int e = 0; e < 16; ++e) {
                int kk = k0 + hi * 8 + ((e < 8) ? e : e + 8);
                af[e] = (kk < K) ? (_Float16)A[abase + kk] : (_Float16)0.0f;
            }
        }

        // ---- B fragment: lane's column ncol, K = k0+[0..7, 16..23]; lanes 16-31 +8
        const int bkb = k0 + hi * 8;
        if (full && alignW && ncol < N) {
            const float* pw = W + (long)ncol * K + bkb;
            float4 w0 = *(const float4*)(pw);
            float4 w1 = *(const float4*)(pw + 4);
            float4 w2 = *(const float4*)(pw + 16);
            float4 w3 = *(const float4*)(pw + 20);
            cvt8(bf, 0, w0, w1);
            cvt8(bf, 8, w2, w3);
        } else {
#pragma unroll
            for (int e = 0; e < 16; ++e) {
                int kk = bkb + ((e < 8) ? e : e + 8);
                bf[e] = (ncol < N && kk < K) ? (_Float16)W[(long)ncol * K + kk]
                                             : (_Float16)0.0f;
            }
        }

        acc = __builtin_amdgcn_wmma_f32_16x16x32_f16(false, af, false, bf,
                                                     (short)0, acc, false, false);
    }

#pragma unroll
    for (int v = 0; v < 8; ++v) {
        if (ncol < N) {
            int m = mT * 16 + v + hi * 8;
            float r = acc[v];
            if (ACT == 1) r = fmaxf(r, 0.0f);
            else if (ACT == 2) r = sigf(r);
            out[(long)m * N + ncol] = r;
        }
    }
}

// ---------------------------------------------------------------------------
// Fused message GEMM: out[e,:] = relu( concat(xn[src[e]], ee[e]) * W^T + b )
// K=128, N=64 fixed. Gather fused into the A-fragment load (saves a 410 MB
// concat buffer). Both gather sources are 64-float aligned rows and every
// 8-element fragment group lies entirely in one source -> all b128 loads.
// ---------------------------------------------------------------------------
__global__ void __launch_bounds__(128) gemm_msg_kernel(
    const float* __restrict__ xn, const float* __restrict__ ee,
    const int* __restrict__ src,
    const float* __restrict__ W, const float* __restrict__ bias,
    float* __restrict__ out, int M)
{
    const int lane = threadIdx.x & 31;
    const int wave = blockIdx.x * 4 + (threadIdx.x >> 5);
    const int mT = wave >> 2;        // nT16 == 4 (N=64)
    const int nT = wave & 3;
    if (mT * 16 >= M) return;

    const int mrow = mT * 16 + (lane & 15);
    const int ncol = nT * 16 + (lane & 15);
    const int hi   = lane >> 4;
    const int s    = src[mrow];
    const float* xrow = xn + (long)s * 64;
    const float* erow = ee + (long)mrow * 64;

    v8f acc;
    float b0 = bias[ncol];
#pragma unroll
    for (int v = 0; v < 8; ++v) acc[v] = b0;

#pragma unroll
    for (int k0 = 0; k0 < 128; k0 += 32) {
        v16h af, bf;
        int g0 = k0 + hi * 8;        // first 8-element group (8-aligned)
        int g1 = g0 + 16;            // second group
        const float* p0 = (g0 < 64) ? (xrow + g0) : (erow + g0 - 64);
        const float* p1 = (g1 < 64) ? (xrow + g1) : (erow + g1 - 64);
        float4 a0 = *(const float4*)(p0);
        float4 a1 = *(const float4*)(p0 + 4);
        float4 a2 = *(const float4*)(p1);
        float4 a3 = *(const float4*)(p1 + 4);
        cvt8(af, 0, a0, a1);
        cvt8(af, 8, a2, a3);

        const float* pw = W + (long)ncol * 128 + k0 + hi * 8;
        float4 w0 = *(const float4*)(pw);
        float4 w1 = *(const float4*)(pw + 4);
        float4 w2 = *(const float4*)(pw + 16);
        float4 w3 = *(const float4*)(pw + 20);
        cvt8(bf, 0, w0, w1);
        cvt8(bf, 8, w2, w3);

        acc = __builtin_amdgcn_wmma_f32_16x16x32_f16(false, af, false, bf,
                                                     (short)0, acc, false, false);
    }
#pragma unroll
    for (int v = 0; v < 8; ++v) {
        int m = mT * 16 + v + hi * 8;
        out[(long)m * 64 + ncol] = fmaxf(acc[v], 0.0f);
    }
}

// ---------------------------------------------------------------------------
// Pointwise / irregular kernels
// ---------------------------------------------------------------------------
__global__ void scatter_sum_kernel(const float* __restrict__ msg,
                                   const int* __restrict__ dst,
                                   float* __restrict__ sum, int E)
{
    long idx = (long)blockIdx.x * blockDim.x + threadIdx.x;
    if (idx >= (long)E * 64) return;
    int e = (int)(idx >> 6), c = (int)(idx & 63);
    atomAddF(&sum[(long)dst[e] * 64 + c], msg[idx]);
}

__global__ void count_kernel(const int* __restrict__ dst, float* __restrict__ cnt, int E)
{
    int e = blockIdx.x * blockDim.x + threadIdx.x;
    if (e >= E) return;
    atomAddF(&cnt[dst[e]], 1.0f);
}

__global__ void agg_finalize_kernel(float* __restrict__ sum,
                                    const float* __restrict__ cnt,
                                    const float* __restrict__ xn, int M)
{
    int idx = blockIdx.x * blockDim.x + threadIdx.x;
    if (idx >= M * 64) return;
    int n = idx >> 6;
    sum[idx] = sum[idx] / fmaxf(cnt[n], 1.0f) + xn[idx];
}

__global__ void bn_reduce_kernel(const float* __restrict__ x,
                                 float* __restrict__ stats, int M)
{
    __shared__ float ss[256], sq[256];
    int c = blockIdx.x;  // one block per channel (64)
    float s = 0.0f, q = 0.0f;
    for (int r = threadIdx.x; r < M; r += 256) {
        float v = x[(long)r * 64 + c];
        s += v; q += v * v;
    }
    ss[threadIdx.x] = s; sq[threadIdx.x] = q;
    __syncthreads();
    for (int st = 128; st > 0; st >>= 1) {
        if (threadIdx.x < st) {
            ss[threadIdx.x] += ss[threadIdx.x + st];
            sq[threadIdx.x] += sq[threadIdx.x + st];
        }
        __syncthreads();
    }
    if (threadIdx.x == 0) {
        float mu = ss[0] / M;
        stats[c] = mu;
        stats[64 + c] = sq[0] / M - mu * mu;
    }
}

__global__ void bn_apply_kernel(float* __restrict__ h, const float* __restrict__ hn,
                                const float* __restrict__ stats,
                                const float* __restrict__ g,
                                const float* __restrict__ b, int M)
{
    int idx = blockIdx.x * blockDim.x + threadIdx.x;
    if (idx >= M * 64) return;
    int c = idx & 63;
    float v = (hn[idx] - stats[c]) * rsqrtf(stats[64 + c] + BN_EPS) * g[c] + b[c];
    h[idx] += fmaxf(v, 0.0f);  // residual + relu
}

__global__ void fill_kernel(float* __restrict__ p, float v, int n)
{
    int i = blockIdx.x * blockDim.x + threadIdx.x;
    if (i < n) p[i] = v;
}

__global__ void rsqrt_kernel(float* __restrict__ d, int n)
{
    int i = blockIdx.x * blockDim.x + threadIdx.x;
    if (i < n) d[i] = (d[i] > 0.0f) ? rsqrtf(d[i]) : 0.0f;
}

__global__ void gcn_init_kernel(const float* __restrict__ xw,
                                const float* __restrict__ dinv,
                                const float* __restrict__ b,
                                float* __restrict__ out, int M)
{
    int idx = blockIdx.x * blockDim.x + threadIdx.x;
    if (idx >= M * 64) return;
    int n = idx >> 6, c = idx & 63;
    out[idx] = xw[idx] * dinv[n] * dinv[n] + b[c];  // self-loop + bias
}

__global__ void gcn_scatter_kernel(const float* __restrict__ xw,
                                   const int* __restrict__ src,
                                   const int* __restrict__ dst,
                                   const float* __restrict__ dinv,
                                   float* __restrict__ out, int E)
{
    long idx = (long)blockIdx.x * blockDim.x + threadIdx.x;
    if (idx >= (long)E * 64) return;
    int e = (int)(idx >> 6), c = (int)(idx & 63);
    int s = src[e], d = dst[e];
    atomAddF(&out[(long)d * 64 + c], xw[(long)s * 64 + c] * dinv[s] * dinv[d]);
}

__global__ void gcn_epilogue_kernel(const float* __restrict__ accum,
                                    float* __restrict__ c, int M, int addres)
{
    int idx = blockIdx.x * blockDim.x + threadIdx.x;
    if (idx >= M * 64) return;
    float v = fmaxf(accum[idx], 0.0f);
    c[idx] = addres ? (c[idx] + v) : v;
}

__global__ void scale_att_kernel(float* __restrict__ c, const float* __restrict__ att, int M)
{
    int idx = blockIdx.x * blockDim.x + threadIdx.x;
    if (idx >= M * 64) return;
    c[idx] *= att[idx >> 6];
}

__global__ void lstm_cell_kernel(const float* __restrict__ g,
                                 float* __restrict__ cst, float* __restrict__ hst,
                                 float* __restrict__ y, int t, int M)
{
    int idx = blockIdx.x * blockDim.x + threadIdx.x;
    if (idx >= M * 64) return;
    int n = idx >> 6, c = idx & 63;
    long gb = (long)n * 256;
    float ig = sigf(g[gb + c]);
    float fg = sigf(g[gb + 64 + c]);
    float gg = tanhf(g[gb + 128 + c]);
    float og = sigf(g[gb + 192 + c]);
    float cn = fg * cst[idx] + ig * gg;
    float hn = og * tanhf(cn);
    cst[idx] = cn;
    hst[idx] = hn;
    y[(long)n * (SEQ * 64) + t * 64 + c] = hn;
}

__global__ void mha_attn_kernel(const float* __restrict__ qkv,
                                float* __restrict__ attno, int nNodes)
{
    int idx = blockIdx.x * blockDim.x + threadIdx.x;
    if (idx >= nNodes * 8) return;
    int n = idx >> 3, h = idx & 7;
    const float scale = 0.3535533905932738f;  // 1/sqrt(8)
    long base = (long)n * SEQ * 192 + h * 8;
    for (int tq = 0; tq < SEQ; ++tq) {
        float q[8];
#pragma unroll
        for (int d = 0; d < 8; ++d) q[d] = qkv[base + (long)tq * 192 + d];
        float s[SEQ], mx = -1e30f;
        for (int tk = 0; tk < SEQ; ++tk) {
            float a = 0.0f;
#pragma unroll
            for (int d = 0; d < 8; ++d) a += q[d] * qkv[base + (long)tk * 192 + 64 + d];
            s[tk] = a * scale;
            mx = fmaxf(mx, s[tk]);
        }
        float den = 0.0f;
        for (int tk = 0; tk < SEQ; ++tk) { s[tk] = expf(s[tk] - mx); den += s[tk]; }
        float inv = 1.0f / den;
        float o[8];
#pragma unroll
        for (int d = 0; d < 8; ++d) o[d] = 0.0f;
        for (int tk = 0; tk < SEQ; ++tk) {
#pragma unroll
            for (int d = 0; d < 8; ++d)
                o[d] += s[tk] * qkv[base + (long)tk * 192 + 128 + d];
        }
        long ob = (long)(n * SEQ + tq) * 64 + h * 8;
#pragma unroll
        for (int d = 0; d < 8; ++d) attno[ob + d] = o[d] * inv;
    }
}

// ---------------------------------------------------------------------------
// Host-side orchestration
// ---------------------------------------------------------------------------
static void launch_gemm(int act, const float* A, long lda, long aoff,
                        const float* W, const float* bias, const float* resid,
                        float* out, int M, int N, int K, hipStream_t s)
{
    int tiles = ((M + 15) >> 4) * ((N + 15) >> 4);
    int blocks = (tiles + 3) >> 2;
    dim3 g(blocks), b(128);
    switch (act) {
        case 0: gemm_wmma_kernel<0><<<g, b, 0, s>>>(A, lda, aoff, W, bias, resid, out, M, N, K); break;
        case 1: gemm_wmma_kernel<1><<<g, b, 0, s>>>(A, lda, aoff, W, bias, resid, out, M, N, K); break;
        default: gemm_wmma_kernel<2><<<g, b, 0, s>>>(A, lda, aoff, W, bias, resid, out, M, N, K); break;
    }
}

#define EW(kern, total, ...)                                                    \
    kern<<<(int)((((long)(total)) + 255) / 256), 256, 0, stream>>>(__VA_ARGS__)

extern "C" void kernel_launch(void* const* d_in, const int* in_sizes, int n_in,
                              void* d_out, int out_size, void* d_ws, size_t ws_size,
                              hipStream_t stream)
{
    (void)in_sizes; (void)n_in; (void)out_size; (void)ws_size;

    const float* x     = (const float*)d_in[0];
    const int*   ei    = (const int*)d_in[1];
    const float* eattr = (const float*)d_in[2];
    const float* tseq  = (const float*)d_in[3];
    const int* src = ei;
    const int* dst = ei + N_EDGES;
    auto P = [&](int i) { return (const float*)d_in[i]; };
    // Param leaf order (pytree / sorted dict keys after the 4 arrays):
    // 4:att1.b 5:att1.w 6:att2.b 7:att2.w 8..13:bn{0..2}.{b,g}
    // 14:cls1.b 15:cls1.w 16:cls2.b 17:cls2.w
    // 18..53: convs[0..2].{e1,e2,m1,m2,n1,n2}.{b,w}
    // 54..59: gcns[0..2].{b,w} 60:inproj.b 61:inproj.w
    // 62..69: lstm[0..1].{b_hh,b_ih,w_hh,w_ih}
    // 70:mha.in_b 71:mha.in_w 72:mha.out_b 73:mha.out_w
    // 74,75:out_mlp1 76,77:out_mlp2 78,79:output_proj 80,81:tp1 82,83:tp2

    float* out = (float*)d_out;
    float* ws  = (float*)d_ws;

    // workspace arena (element offsets, all 16-float aligned for b128 loads)
    float* h     = ws;                 // 3.2M
    float* xn    = ws + 3200000;       // 3.2M
    float* hsum  = ws + 6400000;       // 3.2M
    float* tmp   = ws + 9600000;       // 3.2M
    float* cbuf  = ws + 12800000;      // 3.2M
    float* cnacc = ws + 16000000;      // 3.2M
    float* xw    = ws + 19200000;      // 3.2M
    float* cnt   = ws + 22400000;      // 64k (counts, then deg/dinv)
    float* att   = ws + 22464000;      // 64k
    float* stats = ws + 22528000;      // 128
    float* big   = ws + 22600000;      // shared big region
    float* ehid  = big;                // 51.2M (phase A)
    float* ee    = big + 51200000;     // 51.2M (phase A)
    float* t0    = big;                // 32M   (phase C, reuses phase-A space)
    float* t1    = big + 32000000;     // 32M
    float* qkv   = big + 64000000;     // 96M
    float* attno = big + 160000000;    // 32M
    float* gates = big + 192000000;    // 12.8M
    float* hst   = big + 204800000;    // 3.2M
    float* cst   = big + 208000000;    // 3.2M

    float* out_spatial = out;               // [50000,32]
    float* out_logits  = out + 1600000;     // [50000,2]
    float* out_tf      = out + 1700000;     // [50000,10,32]
    float* out_tp      = out + 17700000;    // [50000,6]

    // ---------------- SpatialGraphNetwork ----------------
    launch_gemm(0, x, IN_DIM, 0, P(61), P(60), nullptr, h, N_NODES, HID, IN_DIM, stream);
    for (int L = 0; L < 3; ++L) {
        int pb = 18 + 12 * L;
        const float *e1b = P(pb),     *e1w = P(pb + 1);
        const float *e2b = P(pb + 2), *e2w = P(pb + 3);
        const float *m1b = P(pb + 4), *m1w = P(pb + 5);
        const float *m2b = P(pb + 6), *m2w = P(pb + 7);
        const float *n1b = P(pb + 8), *n1w = P(pb + 9);
        const float *n2b = P(pb + 10), *n2w = P(pb + 11);
        const float *bnb = P(8 + 2 * L), *bng = P(9 + 2 * L);

        launch_gemm(1, h, HID, 0, n1w, n1b, nullptr, tmp, N_NODES, HID, HID, stream);
        launch_gemm(0, tmp, HID, 0, n2w, n2b, nullptr, xn, N_NODES, HID, HID, stream);
        launch_gemm(1, eattr, 3, 0, e1w, e1b, nullptr, ehid, N_EDGES, HID, 3, stream);
        launch_gemm(0, ehid, HID, 0, e2w, e2b, nullptr, ee, N_EDGES, HID, HID, stream);
        {
            int tiles = (N_EDGES / 16) * 4;
            gemm_msg_kernel<<<(tiles + 3) / 4, 128, 0, stream>>>(xn, ee, src, m1w, m1b,
                                                                 ehid, N_EDGES);
        }
        launch_gemm(0, ehid, HID, 0, m2w, m2b, nullptr, ee, N_EDGES, HID, HID, stream);

        hipMemsetAsync(hsum, 0, sizeof(float) * N_NODES * 64, stream);
        hipMemsetAsync(cnt, 0, sizeof(float) * N_NODES, stream);
        EW(scatter_sum_kernel, (long)N_EDGES * 64, ee, dst, hsum, N_EDGES);
        EW(count_kernel, N_EDGES, dst, cnt, N_EDGES);
        EW(agg_finalize_kernel, N_NODES * 64, hsum, cnt, xn, N_NODES);
        bn_reduce_kernel<<<64, 256, 0, stream>>>(hsum, stats, N_NODES);
        EW(bn_apply_kernel, N_NODES * 64, h, hsum, stats, bng, bnb, N_NODES);
    }
    launch_gemm(0, h, HID, 0, P(79), P(78), nullptr, out_spatial, N_NODES, SP, HID, stream);

    // ---------------- ClutterClassificationGNN ----------------
    EW(fill_kernel, N_NODES, cnt, 1.0f, N_NODES);          // self-loop degree
    EW(count_kernel, N_EDGES, dst, cnt, N_EDGES);
    EW(rsqrt_kernel, N_NODES, cnt, N_NODES);               // cnt := d^{-1/2}
    for (int i = 0; i < 3; ++i) {
        const float* gb = P(54 + 2 * i);
        const float* gw = P(55 + 2 * i);
        const float* Ain = (i == 0) ? (const float*)out_spatial : (const float*)cbuf;
        int K = (i == 0) ? SP : HID;
        launch_gemm(0, Ain, K, 0, gw, nullptr, nullptr, xw, N_NODES, HID, K, stream);
        EW(gcn_init_kernel, N_NODES * 64, xw, cnt, gb, cnacc, N_NODES);
        EW(gcn_scatter_kernel, (long)N_EDGES * 64, xw, src, dst, cnt, cnacc, N_EDGES);
        EW(gcn_epilogue_kernel, N_NODES * 64, cnacc, cbuf, N_NODES, (i > 0) ? 1 : 0);
    }
    launch_gemm(1, cbuf, HID, 0, P(5), P(4), nullptr, tmp, N_NODES, 32, HID, stream);
    launch_gemm(2, tmp, 32, 0, P(7), P(6), nullptr, att, N_NODES, 1, 32, stream);
    EW(scale_att_kernel, N_NODES * 64, cbuf, att, N_NODES);
    launch_gemm(1, cbuf, HID, 0, P(15), P(14), nullptr, tmp, N_NODES, 32, HID, stream);
    launch_gemm(0, tmp, 32, 0, P(17), P(16), nullptr, out_logits, N_NODES, 2, 32, stream);

    // ---------------- TemporalGraphNetwork ----------------
    for (int l = 0; l < 2; ++l) {
        const float* bhh = P(62 + 4 * l);
        const float* bih = P(63 + 4 * l);
        const float* whh = P(64 + 4 * l);
        const float* wih = P(65 + 4 * l);
        hipMemsetAsync(hst, 0, sizeof(float) * N_NODES * HID, stream);
        hipMemsetAsync(cst, 0, sizeof(float) * N_NODES * HID, stream);
        const float* X = (l == 0) ? tseq : (const float*)t0;
        long ldx = (l == 0) ? (long)SEQ * SP : (long)SEQ * HID;
        int Kx = (l == 0) ? SP : HID;
        float* Y = (l == 0) ? t0 : t1;
        for (int t = 0; t < SEQ; ++t) {
            launch_gemm(0, X, ldx, (long)t * Kx, wih, bih, nullptr, gates,
                        N_NODES, 4 * HID, Kx, stream);
            launch_gemm(0, hst, HID, 0, whh, bhh, gates, gates,
                        N_NODES, 4 * HID, HID, stream);
            EW(lstm_cell_kernel, N_NODES * 64, gates, cst, hst, Y, t, N_NODES);
        }
    }
    // MHA
    launch_gemm(0, t1, HID, 0, P(71), P(70), nullptr, qkv, N_NODES * SEQ, 192, HID, stream);
    EW(mha_attn_kernel, N_NODES * 8, qkv, attno, N_NODES);
    // comb = t1 + out_proj(attno)
    launch_gemm(0, attno, HID, 0, P(73), P(72), t1, t0, N_NODES * SEQ, HID, HID, stream);
    // tf = out_mlp2(relu(out_mlp1(comb)))
    launch_gemm(1, t0, HID, 0, P(75), P(74), nullptr, t1, N_NODES * SEQ, HID, HID, stream);
    launch_gemm(0, t1, HID, 0, P(77), P(76), nullptr, out_tf, N_NODES * SEQ, SP, HID, stream);
    // tp = tp2(relu(tp1(tf[:, -1, :])))
    launch_gemm(1, out_tf, (long)SEQ * SP, (long)(SEQ - 1) * SP, P(81), P(80), nullptr,
                tmp, N_NODES, 64, SP, stream);
    launch_gemm(0, tmp, 64, 0, P(83), P(82), nullptr, out_tp, N_NODES, 6, 64, stream);
}
